// CandidateSelector_19232863552247
// MI455X (gfx1250) — compile-verified, matrix operand order checked by reading, not run on hardware
//
#include <hip/hip_runtime.h>
#include <stdint.h>

#define N_OBJ  65536
#define N_FRM  1024
#define OPF    64
#define DIM    256
#define NQ     512
#define MAXSEL 2048
#define BPAD   264     // B-tile row stride in bf16 elems (528B = 132 dwords, mod64=4)

typedef __attribute__((ext_vector_type(16))) __bf16 v16bf;
typedef __attribute__((ext_vector_type(8)))  float  v8f;

union Frag { uint4 u[2]; v16bf v; };

__device__ __forceinline__ uint32_t pack_bf16x2(float a, float b) {
    uint32_t ba = __float_as_uint(a);
    uint32_t bb = __float_as_uint(b);
    uint32_t ra = (ba + 0x7fffu + ((ba >> 16) & 1u)) >> 16;   // RNE
    uint32_t rb = (bb + 0x7fffu + ((bb >> 16) & 1u)) >> 16;
    return (ra & 0xffffu) | (rb << 16);
}

// gfx1250 async copy global -> LDS (tracked by ASYNCcnt). lds_off is the
// byte offset within the workgroup LDS allocation (low 32 bits of the flat
// shared-memory pointer).
__device__ __forceinline__ void async_ld_b128(uint32_t lds_off, uint64_t gaddr) {
    asm volatile("global_load_async_to_lds_b128 %0, %1, off"
                 :: "v"(lds_off), "v"(gaddr) : "memory");
}

// ---------------------------------------------------------------- queries ---
// L2-normalize 512 query rows (D=256) -> bf16, row-major. One wave per row.
__global__ __launch_bounds__(256)
void normalize_q(const float* __restrict__ in, uint16_t* __restrict__ out) {
    int wave = (blockIdx.x * blockDim.x + threadIdx.x) >> 5;
    int lane = threadIdx.x & 31;
    if (wave >= NQ) return;
    const float4* row = (const float4*)(in + (size_t)wave * DIM);
    float4 a = row[lane * 2 + 0];
    float4 b = row[lane * 2 + 1];
    float ss = a.x*a.x + a.y*a.y + a.z*a.z + a.w*a.w
             + b.x*b.x + b.y*b.y + b.z*b.z + b.w*b.w;
    #pragma unroll
    for (int off = 16; off > 0; off >>= 1) ss += __shfl_xor(ss, off, 32);
    float inv = 1.0f / fmaxf(sqrtf(ss), 1e-12f);
    uint4 st;
    st.x = pack_bf16x2(a.x * inv, a.y * inv);
    st.y = pack_bf16x2(a.z * inv, a.w * inv);
    st.z = pack_bf16x2(b.x * inv, b.y * inv);
    st.w = pack_bf16x2(b.z * inv, b.w * inv);
    ((uint4*)(out + (size_t)wave * DIM))[lane] = st;
}

// ------------------------------------------------------------------- GEMM ---
// 256 threads = 8 waves; each wave owns a 16-row object tile (block = 128
// rows). Block loops over 32 query tiles; each 16x256 bf16 B tile is staged
// in double-buffered LDS via global_load_async_to_lds_b128 and consumed by
// all 8 waves through ds_load_b128 feeding v_wmma_f32_16x16x32_bf16.
__global__ __launch_bounds__(256)
void cos_gemm_wmma(const float* __restrict__ obj,
                   const uint16_t* __restrict__ qnb,
                   const int* __restrict__ mask,
                   float* __restrict__ scores_out,   // d_out[0..N_OBJ)
                   int*   __restrict__ rowarg) {
    __shared__ __align__(16) uint16_t Bt[2][16][BPAD];

    const int tid  = threadIdx.x;
    const int wave = tid >> 5;
    const int lane = tid & 31;
    const int half = lane >> 4;          // K-half select (A/B 16-bit layout)
    const int l15  = lane & 15;
    const int m0   = blockIdx.x * 128 + wave * 16;
    const int arow = m0 + l15;
    const float* ap = obj + (size_t)arow * DIM;
    __builtin_prefetch(ap, 0, 1);        // global_prefetch_b8

    // this thread's two b128 chunks of each B tile (tile = 512 chunks of 8 bf16)
    const int ci0 = tid, ci1 = tid + 256;
    const int r0 = ci0 >> 5, c0 = (ci0 & 31) * 8;
    const int r1 = ci1 >> 5, c1 = (ci1 & 31) * 8;

    // prologue: async-stage B tile 0 into buffer 0
    async_ld_b128((uint32_t)(uintptr_t)&Bt[0][r0][c0],
                  (uint64_t)(uintptr_t)(qnb + (size_t)r0 * DIM + c0));
    async_ld_b128((uint32_t)(uintptr_t)&Bt[0][r1][c1],
                  (uint64_t)(uintptr_t)(qnb + (size_t)r1 * DIM + c1));

    // pass 1: sum of squares over this lane's 128 K-values of row `arow`
    float ss = 0.0f;
    #pragma unroll
    for (int ks = 0; ks < 8; ++ks) {
        int kb = ks * 32 + half * 8;
        float4 x0 = *(const float4*)(ap + kb);
        float4 x1 = *(const float4*)(ap + kb + 4);
        float4 x2 = *(const float4*)(ap + kb + 16);
        float4 x3 = *(const float4*)(ap + kb + 20);
        ss += x0.x*x0.x + x0.y*x0.y + x0.z*x0.z + x0.w*x0.w;
        ss += x1.x*x1.x + x1.y*x1.y + x1.z*x1.z + x1.w*x1.w;
        ss += x2.x*x2.x + x2.y*x2.y + x2.z*x2.z + x2.w*x2.w;
        ss += x3.x*x3.x + x3.y*x3.y + x3.z*x3.z + x3.w*x3.w;
    }
    ss += __shfl_xor(ss, 16, 32);        // lanes l and l+16 hold same row
    float inv = 1.0f / fmaxf(sqrtf(ss), 1e-12f);

    // pass 2: re-read (WGP$-hot), normalize, pack into A fragments
    Frag afr[8];
    #pragma unroll
    for (int ks = 0; ks < 8; ++ks) {
        int kb = ks * 32 + half * 8;
        float4 x0 = *(const float4*)(ap + kb);
        float4 x1 = *(const float4*)(ap + kb + 4);
        float4 x2 = *(const float4*)(ap + kb + 16);
        float4 x3 = *(const float4*)(ap + kb + 20);
        afr[ks].u[0] = make_uint4(pack_bf16x2(x0.x*inv, x0.y*inv),
                                  pack_bf16x2(x0.z*inv, x0.w*inv),
                                  pack_bf16x2(x1.x*inv, x1.y*inv),
                                  pack_bf16x2(x1.z*inv, x1.w*inv));
        afr[ks].u[1] = make_uint4(pack_bf16x2(x2.x*inv, x2.y*inv),
                                  pack_bf16x2(x2.z*inv, x2.w*inv),
                                  pack_bf16x2(x3.x*inv, x3.y*inv),
                                  pack_bf16x2(x3.z*inv, x3.w*inv));
    }

    float rmax[8]; int rarg[8];
    #pragma unroll
    for (int r = 0; r < 8; ++r) { rmax[r] = -3.0e38f; rarg[r] = 0; }

    for (int qt = 0; qt < 32; ++qt) {
        const int buf = qt & 1;
        if (qt + 1 < 32) {               // stream next tile into other buffer
            const uint16_t* nq = qnb + (size_t)(qt + 1) * 16 * DIM;
            async_ld_b128((uint32_t)(uintptr_t)&Bt[buf ^ 1][r0][c0],
                          (uint64_t)(uintptr_t)(nq + (size_t)r0 * DIM + c0));
            async_ld_b128((uint32_t)(uintptr_t)&Bt[buf ^ 1][r1][c1],
                          (uint64_t)(uintptr_t)(nq + (size_t)r1 * DIM + c1));
            asm volatile("s_wait_asynccnt 2" ::: "memory");
        } else {
            asm volatile("s_wait_asynccnt 0" ::: "memory");
        }
        __syncthreads();                 // current buffer fully staged

        v8f c = {};
        #pragma unroll
        for (int ks = 0; ks < 8; ++ks) {
            int kb = ks * 32 + half * 8;
            Frag bfr;
            bfr.u[0] = *(const uint4*)&Bt[buf][l15][kb];
            bfr.u[1] = *(const uint4*)&Bt[buf][l15][kb + 16];
            c = __builtin_amdgcn_wmma_f32_16x16x32_bf16(
                    false, afr[ks].v, false, bfr.v, (short)0, c, false, false);
        }
        const int q0 = qt * 16;
        #pragma unroll
        for (int r = 0; r < 8; ++r) {
            float s = c[r];
            if (s > rmax[r]) { rmax[r] = s; rarg[r] = q0 + l15; }
        }
        __syncthreads();                 // done reading before buffer reuse
    }

    // reduce (max, first-argmax) over the 16 lanes of each C half
    #pragma unroll
    for (int r = 0; r < 8; ++r) {
        float m = rmax[r]; int a = rarg[r];
        #pragma unroll
        for (int off = 1; off < 16; off <<= 1) {
            float om = __shfl_xor(m, off, 16);
            int   oa = __shfl_xor(a, off, 16);
            if (om > m || (om == m && oa < a)) { m = om; a = oa; }
        }
        rmax[r] = m; rarg[r] = a;
    }
    if (l15 == 0) {
        #pragma unroll
        for (int r = 0; r < 8; ++r) {
            int row = m0 + half * 8 + r;       // lanes>=16 hold M = r+8
            float s = (mask[row] == 1) ? rmax[r] : 0.0f;
            scores_out[row] = s;
            rowarg[row]     = rarg[r];
        }
    }
}

// ------------------------------------------------------- frame-max select ---
__global__ __launch_bounds__(1024)
void frame_select(const float* __restrict__ scores,
                  float* __restrict__ selS, int* __restrict__ selI,
                  int* __restrict__ nsel) {
    __shared__ int offs[N_FRM];
    int f = threadIdx.x;
    const float* s = scores + f * OPF;
    float fm = s[0];
    for (int i = 1; i < OPF; ++i) fm = fmaxf(fm, s[i]);
    int c = 0;
    for (int i = 0; i < OPF; ++i) if (s[i] >= fm) ++c;
    offs[f] = c;
    __syncthreads();
    for (int d = 1; d < N_FRM; d <<= 1) {          // inclusive scan
        int t = (f >= d) ? offs[f - d] : 0;
        __syncthreads();
        offs[f] += t;
        __syncthreads();
    }
    int incl = offs[f];
    int excl = incl - c;
    if (f == N_FRM - 1) *nsel = (incl > MAXSEL) ? MAXSEL : incl;
    int o = excl;
    for (int i = 0; i < OPF && o < MAXSEL; ++i)
        if (s[i] >= fm) { selS[o] = s[i]; selI[o] = f * OPF + i; ++o; }
}

// --------------------------------------------------------------- finalize ---
// single wave: qfn suppression -> greedy NMS -> top-k -> top-p -> outputs
__global__ __launch_bounds__(32)
void finalize(const float* __restrict__ selS, const int* __restrict__ selI,
              const int* __restrict__ nselp, const int* __restrict__ qarg,
              const int* __restrict__ qfnp, const int* __restrict__ topkp,
              const float* __restrict__ toppp, const float* __restrict__ thrp,
              float* __restrict__ dout) {
    __shared__ float sS[MAXSEL];   // working scores (mutated by NMS)
    __shared__ float sV[MAXSEL];   // post-suppression values (frozen)
    __shared__ int   sK[MAXSEL];   // keep flags
    __shared__ float sT[MAXSEL];   // top-k scratch
    int lane = threadIdx.x;
    int n = *nselp; if (n > MAXSEL) n = MAXSEL; if (n < 0) n = 0;

    for (int i = lane; i < n; i += 32) { sS[i] = selS[i]; sK[i] = 0; }
    __syncthreads();

    if (lane == 0 && n > 0) {                      // query-frame suppression
        int qpos = *qfnp;
        if (qpos >= n) qpos = n - 1;
        for (int j = qpos; j >= 0 && sS[j] >= 0.8f; --j) sS[j] = 0.0f;
    }
    __syncthreads();
    for (int i = lane; i < n; i += 32) sV[i] = sS[i];
    __syncthreads();

    float thr = *thrp;
    if (thr > 0.0f) {
        for (;;) {
            float bm = -1.0f; int bi = -1;
            for (int i = lane; i < n; i += 32)
                if (sS[i] > bm) { bm = sS[i]; bi = i; }
            #pragma unroll
            for (int off = 16; off > 0; off >>= 1) {
                float om = __shfl_xor(bm, off, 32);
                int   oi = __shfl_xor(bi, off, 32);
                if (om > bm || (om == bm && oi >= 0 && oi < bi)) { bm = om; bi = oi; }
            }
            if (bm <= 0.0f) break;
            if (lane == 0) {
                float val = thr * bm;
                sK[bi] = 1;
                sS[bi] = 0.0f;
                for (int j = bi - 1; j >= 0 && sS[j] > val; --j) sS[j] = 0.0f;
                for (int j = bi + 1; j <  n && sS[j] > val; ++j) sS[j] = 0.0f;
            }
            __syncthreads();
        }
    } else {
        for (int i = lane; i < n; i += 32) sK[i] = 1;
    }
    __syncthreads();

    if (lane == 0) {
        int k = *topkp;
        int nk = 0;
        for (int i = 0; i < n; ++i) if (sK[i]) sT[nk++] = sV[i];
        float thrk = -3.0e38f;
        if (k < nk) {                              // k-th largest (ties kept)
            for (int it = 0; it < k; ++it) {
                int bj = 0; float bv = -3.0e38f;
                for (int j = 0; j < nk; ++j)
                    if (sT[j] > bv) { bv = sT[j]; bj = j; }
                thrk = bv; sT[bj] = -3.0e38f;
            }
        }
        float mx = -3.0e38f;
        for (int i = 0; i < n; ++i)
            if (sK[i] && sV[i] >= thrk) mx = fmaxf(mx, sV[i]);
        float thp = fminf(*toppp, mx);

        int cnt = 0;
        for (int i = 0; i < n; ++i)
            if (sK[i] && sV[i] >= thrk && sV[i] >= thp) ++cnt;
        int o = 0;
        for (int i = 0; i < n; ++i)
            if (sK[i] && sV[i] >= thrk && sV[i] >= thp) {
                dout[N_OBJ + o]            = sV[i];
                dout[N_OBJ + cnt + o]      = (float)selI[i];
                dout[N_OBJ + 2*cnt + o]    = (float)qarg[selI[i]];
                ++o;
            }
    }
}

// ------------------------------------------------------------------ entry ---
extern "C" void kernel_launch(void* const* d_in, const int* in_sizes, int n_in,
                              void* d_out, int out_size, void* d_ws, size_t ws_size,
                              hipStream_t stream) {
    const float* obj  = (const float*)d_in[0];
    const float* qry  = (const float*)d_in[1];
    const int*   mask = (const int*)  d_in[2];
    /* d_in[3] frame_ids: implicit (64 contiguous per frame) */
    const int*   qfn  = (const int*)  d_in[4];
    const int*   topk = (const int*)  d_in[5];
    const float* topp = (const float*)d_in[6];
    const float* nthr = (const float*)d_in[7];
    float* out = (float*)d_out;

    char* ws = (char*)d_ws;
    uint16_t* qnb    = (uint16_t*)(ws);                     // 512*256*2 = 256KB
    int*      rowarg = (int*)     (ws + (256u << 10));      // 65536*4   = 256KB
    float*    selS   = (float*)   (ws + (512u << 10));      // 8KB
    int*      selI   = (int*)     (ws + (512u << 10) + 8192);
    int*      nsel   = (int*)     (ws + (512u << 10) + 16384);

    normalize_q  <<<NQ / 8, 256, 0, stream>>>(qry, qnb);
    cos_gemm_wmma<<<N_OBJ / 128, 256, 0, stream>>>(obj, qnb, mask, out, rowarg);
    frame_select <<<1, N_FRM, 0, stream>>>(out, selS, selI, nsel);
    finalize     <<<1, 32, 0, stream>>>(selS, selI, nsel, rowarg,
                                        qfn, topk, topp, nthr, out);
}
